// MSARowAttentionWithPairBias_71227737637046
// MI455X (gfx1250) — matrix operand order repaired and checked by reading, not compiled, not run on hardware
//
#include <hip/hip_runtime.h>

#define S_DIM 128
#define I_DIM 256
#define CM    256
#define CZ    128
#define NH    8
#define CH    32
#define HC    256   // NH*CH
#define MROWS (S_DIM * I_DIM)   // 32768

typedef __attribute__((ext_vector_type(16))) _Float16 v16h;
typedef __attribute__((ext_vector_type(8)))  _Float16 h8;
typedef __attribute__((ext_vector_type(8)))  float    v8f;

union V16 { v16h v; h8 h[2]; };

__device__ __forceinline__ float red32(float v) {
#pragma unroll
  for (int off = 16; off >= 1; off >>= 1) v += __shfl_xor(v, off, 32);
  return v;
}

// ---------------------------------------------------------------------------
// Pack a 256x256 f32 weight (row-major K x N) into WMMA 32x16 B-fragment
// layout (f16): frag (kt,nt) holds 512 halves; lane L element e =
// W[32*kt + 8*(L>>4) + (e&7) + 16*(e>>3)][nt*16 + (L&15)].
// ---------------------------------------------------------------------------
__global__ void pack_w_kernel(const float* __restrict__ w0, const float* __restrict__ w1,
                              const float* __restrict__ w2, const float* __restrict__ w3,
                              const float* __restrict__ w4, _Float16* __restrict__ dst) {
  const float* src;
  switch (blockIdx.y) {
    case 0: src = w0; break;
    case 1: src = w1; break;
    case 2: src = w2; break;
    case 3: src = w3; break;
    default: src = w4; break;
  }
  _Float16* d = dst + (size_t)blockIdx.y * 65536;
  int t = blockIdx.x * 256 + threadIdx.x;          // 0..65535
  int frag = t >> 9, r = t & 511;
  int L = r >> 4, e = r & 15;
  int kt = frag >> 4, nt = frag & 15;
  int K = 32 * kt + 8 * (L >> 4) + (e & 7) + 16 * (e >> 3);
  int N = nt * 16 + (L & 15);
  d[t] = (_Float16)src[K * 256 + N];
}

// ---------------------------------------------------------------------------
// LayerNorm of m over C_M=256, f32 -> f16. One wave per row.
// ---------------------------------------------------------------------------
__global__ void ln_m_kernel(const float* __restrict__ m, const float* __restrict__ w,
                            const float* __restrict__ b, _Float16* __restrict__ mn) {
  int wave = threadIdx.x >> 5, lane = threadIdx.x & 31;
  int row = blockIdx.x * 8 + wave;
  const float* src = m + (size_t)row * CM;
  float x[8];
  float s = 0.f;
#pragma unroll
  for (int u = 0; u < 8; ++u) { x[u] = src[lane + 32 * u]; s += x[u]; }
  float mu = red32(s) * (1.f / CM);
  float var = 0.f;
#pragma unroll
  for (int u = 0; u < 8; ++u) { float d = x[u] - mu; var += d * d; }
  var = red32(var) * (1.f / CM);
  float inv = rsqrtf(var + 1e-5f);
  _Float16* dst = mn + (size_t)row * CM;
#pragma unroll
  for (int u = 0; u < 8; ++u) {
    int c = lane + 32 * u;
    dst[c] = (_Float16)((x[u] - mu) * inv * w[c] + b[c]);
  }
}

// ---------------------------------------------------------------------------
// LayerNorm of z over C_Z=128, then (1x128)@Wb(128x8) -> biasT[h][i][j] (f32).
// One wave per (i,j) row.
// ---------------------------------------------------------------------------
__global__ void ln_z_bias_kernel(const float* __restrict__ z, const float* __restrict__ w,
                                 const float* __restrict__ b, const float* __restrict__ Wb,
                                 float* __restrict__ biasT) {
  int wave = threadIdx.x >> 5, lane = threadIdx.x & 31;
  int row = blockIdx.x * 8 + wave;                 // 0..65535 = i*I + j
  const float* src = z + (size_t)row * CZ;
  float x[4];
  float s = 0.f;
#pragma unroll
  for (int u = 0; u < 4; ++u) { x[u] = src[lane + 32 * u]; s += x[u]; }
  float mu = red32(s) * (1.f / CZ);
  float var = 0.f;
#pragma unroll
  for (int u = 0; u < 4; ++u) { float d = x[u] - mu; var += d * d; }
  var = red32(var) * (1.f / CZ);
  float inv = rsqrtf(var + 1e-5f);
  float zn[4];
#pragma unroll
  for (int u = 0; u < 4; ++u) {
    int c = lane + 32 * u;
    zn[u] = (x[u] - mu) * inv * w[c] + b[c];
  }
#pragma unroll
  for (int hh = 0; hh < NH; ++hh) {
    float p = 0.f;
#pragma unroll
    for (int u = 0; u < 4; ++u) p += zn[u] * Wb[(lane + 32 * u) * NH + hh];
    p = red32(p);
    if (lane == 0) biasT[(size_t)hh * I_DIM * I_DIM + row] = p;
  }
}

// ---------------------------------------------------------------------------
// Fused 4-way projection: {q,k,v,gate} = mn @ {Wq,Wk,Wv,Wg}. Each A-fragment
// load feeds FOUR v_wmma_f32_16x16x32_f16 (one per packed weight), 4x'ing
// arithmetic intensity on the activation stream. One 16x16 tile (per output)
// per wave; K-loop of 8.
// ---------------------------------------------------------------------------
__global__ void proj4_kernel(const _Float16* __restrict__ A, const _Float16* __restrict__ Wp,
                             const float* __restrict__ bg, _Float16* __restrict__ qo,
                             _Float16* __restrict__ ko, _Float16* __restrict__ vo,
                             _Float16* __restrict__ go) {
  int wave = threadIdx.x >> 5, lane = threadIdx.x & 31;
  int tile = blockIdx.x * 4 + wave;                // 0..32767
  int mt = tile >> 4, nt = tile & 15;
  int mrow = lane & 15, khalf = lane >> 4;
  v8f acc0 = {}, acc1 = {}, acc2 = {}, acc3 = {};
#pragma unroll
  for (int kt = 0; kt < 8; ++kt) {
    const _Float16* ap = A + (size_t)(mt * 16 + mrow) * CM + 32 * kt + 8 * khalf;
    V16 a; a.h[0] = *(const h8*)ap; a.h[1] = *(const h8*)(ap + 16);
    const _Float16* bp = Wp + (((size_t)kt * 16 + nt) << 9) + lane * 16;
    V16 b0; b0.h[0] = *(const h8*)bp;            b0.h[1] = *(const h8*)(bp + 8);
    V16 b1; b1.h[0] = *(const h8*)(bp + 65536);  b1.h[1] = *(const h8*)(bp + 65536 + 8);
    V16 b2; b2.h[0] = *(const h8*)(bp + 131072); b2.h[1] = *(const h8*)(bp + 131072 + 8);
    V16 b3; b3.h[0] = *(const h8*)(bp + 196608); b3.h[1] = *(const h8*)(bp + 196608 + 8);
    acc0 = __builtin_amdgcn_wmma_f32_16x16x32_f16(false, a.v, false, b0.v, (short)0, acc0, false, false);
    acc1 = __builtin_amdgcn_wmma_f32_16x16x32_f16(false, a.v, false, b1.v, (short)0, acc1, false, false);
    acc2 = __builtin_amdgcn_wmma_f32_16x16x32_f16(false, a.v, false, b2.v, (short)0, acc2, false, false);
    acc3 = __builtin_amdgcn_wmma_f32_16x16x32_f16(false, a.v, false, b3.v, (short)0, acc3, false, false);
  }
  int colg = nt * 16 + (lane & 15);
  float bgv = bg[colg];
#pragma unroll
  for (int r = 0; r < 8; ++r) {
    size_t idx = (size_t)(mt * 16 + r + 8 * khalf) * CM + colg;
    qo[idx] = (_Float16)acc0[r];
    ko[idx] = (_Float16)acc1[r];
    vo[idx] = (_Float16)acc2[r];
    float xx = acc3[r] + bgv;
    go[idx] = (_Float16)(1.f / (1.f + __expf(-xx)));
  }
}

// ---------------------------------------------------------------------------
// Output projection: out(f32) = A(f16) @ WoPack + bo.
// ---------------------------------------------------------------------------
__global__ void gemm_out_kernel(const _Float16* __restrict__ A, const _Float16* __restrict__ Bp,
                                const float* __restrict__ bias, float* __restrict__ outF) {
  int wave = threadIdx.x >> 5, lane = threadIdx.x & 31;
  int tile = blockIdx.x * 4 + wave;
  int mt = tile >> 4, nt = tile & 15;
  int mrow = lane & 15, khalf = lane >> 4;
  v8f acc = {};
#pragma unroll
  for (int kt = 0; kt < 8; ++kt) {
    const _Float16* ap = A + (size_t)(mt * 16 + mrow) * CM + 32 * kt + 8 * khalf;
    V16 a; a.h[0] = *(const h8*)ap; a.h[1] = *(const h8*)(ap + 16);
    const _Float16* bp = Bp + (((size_t)kt * 16 + nt) << 9) + lane * 16;
    V16 bb; bb.h[0] = *(const h8*)bp; bb.h[1] = *(const h8*)(bp + 8);
    acc = __builtin_amdgcn_wmma_f32_16x16x32_f16(false, a.v, false, bb.v, (short)0, acc, false, false);
  }
  int colg = nt * 16 + (lane & 15);
  float bv = bias[colg];
#pragma unroll
  for (int r = 0; r < 8; ++r) {
    outF[(size_t)(mt * 16 + r + 8 * khalf) * CM + colg] = acc[r] + bv;
  }
}

// ---------------------------------------------------------------------------
// Fused attention per (s, head): QK^T (1 WMMA / tile, c=32), scale + pair
// bias + key mask, softmax over j (VGPR-resident 16x256 rows per wave,
// half-wave shuffle reductions), probs->LDS relayout, A@V (16 WMMAs per
// c-tile, V transposed in LDS), gate, f16 out.
// Block = 4 waves; wave handles 4 i-tiles of 16 rows. global_prefetch_b8
// hides the next i-tile's Q/bias latency behind current-tile WMMA+softmax.
// ---------------------------------------------------------------------------
__global__ void attn_kernel(const _Float16* __restrict__ q, const _Float16* __restrict__ k,
                            const _Float16* __restrict__ v, const _Float16* __restrict__ gate,
                            const float* __restrict__ biasT, const float* __restrict__ mask,
                            _Float16* __restrict__ og) {
  __shared__ __attribute__((aligned(16))) _Float16 Vt[CH * I_DIM];       // [c][j], 16 KB
  __shared__ __attribute__((aligned(16))) _Float16 Ps[4 * 16 * I_DIM];   // per-wave probs, 32 KB
  int s = blockIdx.x, hh = blockIdx.y;
  int tid = threadIdx.x, wave = tid >> 5, lane = tid & 31;
  int mrow = lane & 15, khalf = lane >> 4;
  size_t base = ((size_t)s * I_DIM) * HC + (size_t)hh * CH;  // + i*HC + c

  // Stage V transposed into LDS: Vt[c][j]
  for (int t = tid; t < (I_DIM * CH) / 8; t += 128) {
    int j = t >> 2, cc = (t & 3) * 8;
    h8 val = *(const h8*)(v + base + (size_t)j * HC + cc);
#pragma unroll
    for (int e = 0; e < 8; ++e) Vt[(cc + e) * I_DIM + j] = val[e];
  }
  __syncthreads();

  float madd[16];
#pragma unroll
  for (int jt = 0; jt < 16; ++jt)
    madd[jt] = 1e9f * (mask[s * I_DIM + jt * 16 + (lane & 15)] - 1.f);

  const float scale = 0.17677669529663687f;  // 1/sqrt(32)
  const float* bT = biasT + (size_t)hh * I_DIM * I_DIM;

  for (int it = wave; it < 16; it += 4) {
    int i0 = it * 16;
    // Prefetch next i-tile for this wave (Q rows + bias rows) while we crunch
    if (it + 4 < 16) {
      __builtin_prefetch(q + base + (size_t)(i0 + 64 + mrow) * HC, 0, 0);
      __builtin_prefetch(bT + (i0 + 64 + mrow) * I_DIM, 0, 0);
    }
    // Q A-fragment (16x32): two 16B chunks per lane
    const _Float16* qp = q + base + (size_t)(i0 + mrow) * HC + 8 * khalf;
    V16 qa; qa.h[0] = *(const h8*)qp; qa.h[1] = *(const h8*)(qp + 16);

    v8f sc[16];
#pragma unroll
    for (int jt = 0; jt < 16; ++jt) {
      // K^T B-fragment (32x16): two 8-half chunks of one key row
      const _Float16* kp = k + base + (size_t)(jt * 16 + (lane & 15)) * HC + 8 * khalf;
      V16 kb; kb.h[0] = *(const h8*)kp; kb.h[1] = *(const h8*)(kp + 16);
      v8f zc = {};
      sc[jt] = __builtin_amdgcn_wmma_f32_16x16x32_f16(false, qa.v, false, kb.v, (short)0,
                                                      zc, false, false);
    }
    // scale + pair bias + key mask
#pragma unroll
    for (int jt = 0; jt < 16; ++jt) {
      int j = jt * 16 + (lane & 15);
#pragma unroll
      for (int r = 0; r < 8; ++r) {
        int i = i0 + r + 8 * khalf;
        sc[jt][r] = sc[jt][r] * scale + bT[i * I_DIM + j] + madd[jt];
      }
    }
    // softmax over j: rows live across 16 tiles x 16 lanes (half-wave)
#pragma unroll
    for (int r = 0; r < 8; ++r) {
      float mv = -3.0e38f;
#pragma unroll
      for (int jt = 0; jt < 16; ++jt) mv = fmaxf(mv, sc[jt][r]);
#pragma unroll
      for (int off = 8; off >= 1; off >>= 1) mv = fmaxf(mv, __shfl_xor(mv, off, 16));
      float sv = 0.f;
#pragma unroll
      for (int jt = 0; jt < 16; ++jt) {
        float ev = __expf(sc[jt][r] - mv);
        sc[jt][r] = ev;
        sv += ev;
      }
#pragma unroll
      for (int off = 8; off >= 1; off >>= 1) sv += __shfl_xor(sv, off, 16);
      float inv = 1.f / sv;
#pragma unroll
      for (int jt = 0; jt < 16; ++jt) sc[jt][r] *= inv;
    }
    // D-tile -> A-fragment relayout through this wave's LDS strip
    _Float16* P = Ps + wave * (16 * I_DIM);
#pragma unroll
    for (int jt = 0; jt < 16; ++jt)
#pragma unroll
      for (int r = 0; r < 8; ++r)
        P[(r + 8 * khalf) * I_DIM + jt * 16 + (lane & 15)] = (_Float16)sc[jt][r];

    // A @ V : K = j = 256 -> 16 WMMAs per 16-wide c-tile
#pragma unroll
    for (int ct = 0; ct < 2; ++ct) {
      v8f oacc = {};
#pragma unroll
      for (int jt = 0; jt < 16; ++jt) {
        const _Float16* pp = P + mrow * I_DIM + 32 * jt + 8 * khalf;
        V16 pa; pa.h[0] = *(const h8*)pp; pa.h[1] = *(const h8*)(pp + 16);
        const _Float16* vp = Vt + (ct * 16 + (lane & 15)) * I_DIM + 32 * jt + 8 * khalf;
        V16 vb; vb.h[0] = *(const h8*)vp; vb.h[1] = *(const h8*)(vp + 16);
        oacc = __builtin_amdgcn_wmma_f32_16x16x32_f16(false, pa.v, false, vb.v, (short)0,
                                                      oacc, false, false);
      }
#pragma unroll
      for (int r = 0; r < 8; ++r) {
        int i = i0 + r + 8 * khalf;
        int c = ct * 16 + (lane & 15);
        size_t idx = base + (size_t)i * HC + c;
        float g = (float)gate[idx];
        og[idx] = (_Float16)(g * oacc[r]);
      }
    }
  }
}

// ---------------------------------------------------------------------------

extern "C" void kernel_launch(void* const* d_in, const int* in_sizes, int n_in,
                              void* d_out, int out_size, void* d_ws, size_t ws_size,
                              hipStream_t stream) {
  const float* m     = (const float*)d_in[0];
  const float* z     = (const float*)d_in[1];
  const float* mask  = (const float*)d_in[2];
  const float* ln_w  = (const float*)d_in[3];
  const float* ln_b  = (const float*)d_in[4];
  const float* lnz_w = (const float*)d_in[5];
  const float* lnz_b = (const float*)d_in[6];
  const float* Wq    = (const float*)d_in[7];
  const float* Wk    = (const float*)d_in[8];
  const float* Wv    = (const float*)d_in[9];
  const float* Wb    = (const float*)d_in[10];
  const float* Wg    = (const float*)d_in[11];
  const float* bg    = (const float*)d_in[12];
  const float* Wo    = (const float*)d_in[13];
  const float* bo    = (const float*)d_in[14];
  float* out = (float*)d_out;

  char* ws = (char*)d_ws;
  size_t off = 0;
  auto carve = [&](size_t bytes) {
    char* p = ws + off;
    off += (bytes + 255) & ~(size_t)255;
    return p;
  };
  const size_t actBytes = (size_t)MROWS * CM * sizeof(_Float16);  // 16 MB each
  _Float16* mn    = (_Float16*)carve(actBytes);
  _Float16* qbuf  = (_Float16*)carve(actBytes);
  _Float16* kbuf  = (_Float16*)carve(actBytes);
  _Float16* vbuf  = (_Float16*)carve(actBytes);
  _Float16* gbuf  = (_Float16*)carve(actBytes);
  _Float16* obuf  = (_Float16*)carve(actBytes);
  float*    biasT = (float*)carve((size_t)NH * I_DIM * I_DIM * sizeof(float));
  _Float16* wpack = (_Float16*)carve((size_t)5 * 65536 * sizeof(_Float16));
  (void)ws_size; (void)in_sizes; (void)n_in; (void)out_size;

  // 0) pack weights into B-fragment layout (order: Wq, Wk, Wv, Wg, Wo)
  pack_w_kernel<<<dim3(256, 5), 256, 0, stream>>>(Wq, Wk, Wv, Wg, Wo, wpack);
  // 1) LayerNorm(m) -> f16
  ln_m_kernel<<<MROWS / 8, 256, 0, stream>>>(m, ln_w, ln_b, mn);
  // 2) LayerNorm(z) @ Wb -> biasT[h][i][j]
  ln_z_bias_kernel<<<(I_DIM * I_DIM) / 8, 256, 0, stream>>>(z, lnz_w, lnz_b, Wb, biasT);
  // 3) fused q/k/v/gate projection (A-fragment loaded once per 4 WMMAs)
  const int gemmBlocks = (MROWS / 16) * 16 / 4;  // 8192
  proj4_kernel<<<gemmBlocks, 128, 0, stream>>>(mn, wpack, bg, qbuf, kbuf, vbuf, gbuf);
  // 4) fused attention (per s, head)
  attn_kernel<<<dim3(S_DIM, NH), 128, 0, stream>>>(qbuf, kbuf, vbuf, gbuf, biasT, mask, obuf);
  // 5) output projection: out = obuf @ Wo + bo (f32)
  gemm_out_kernel<<<gemmBlocks, 128, 0, stream>>>(obuf, wpack + 4 * 65536, bo, out);
}
